// EncoderGRU_15393162789296
// MI455X (gfx1250) — matrix operand order repaired and checked by reading, not compiled, not run on hardware
//
#include <hip/hip_runtime.h>

typedef _Float16 h16 __attribute__((ext_vector_type(16)));
typedef _Float16 h2  __attribute__((ext_vector_type(2)));
typedef float    f8  __attribute__((ext_vector_type(8)));
typedef float    f4  __attribute__((ext_vector_type(4)));
typedef float    f2  __attribute__((ext_vector_type(2)));
typedef int      i8v __attribute__((ext_vector_type(8)));

#define T_STEPS 512

// Workspace / LDS layout (bytes)
#define WHH_OFF   0         // w_hh A-layout f16: [12 tiles][2 kchunks][32 lanes][16 halves] = 24576 B
#define WIH_OFF   24576     // w_ih A-layout f32 (16x4): [12 tiles][32 lanes][2 f32]        = 3072 B
#define BIASC_OFF 27648     // C-init biases:      [12 tiles][32 lanes][8 f32]              = 12288 B
#define BIASN_OFF 39936     // b_hh for n-gate:    [4 tiles][32 lanes][8 f32]               = 4096 B
#define WIHR_OFF  44032     // raw w_ih [192][4] f32 (fallback path)                        = 3072 B
#define WS_BYTES  47104
#define WS_DW     (WS_BYTES/4)

#if defined(__has_builtin)
#  if __has_builtin(__builtin_amdgcn_wmma_f32_16x16x4_f32)
#    define HAVE_X4 1
#  else
#    define HAVE_X4 0
#  endif
#else
#  define HAVE_X4 0
#endif

__device__ __forceinline__ float fast_tanh(float x) {
#if defined(__has_builtin) && __has_builtin(__builtin_amdgcn_tanhf)
  return __builtin_amdgcn_tanhf(x);     // v_tanh_f32 (CDNA5 TRANS op)
#else
  return tanhf(x);
#endif
}
__device__ __forceinline__ float fast_sigmoid(float x) {
  return 0.5f + 0.5f * fast_tanh(0.5f * x);
}
__device__ __forceinline__ f8 wmma_h(h16 a, h16 b, f8 c) {
  // D(f32 16x16) = A(f16 16x32) x B(f16 32x16) + C
  return __builtin_amdgcn_wmma_f32_16x16x32_f16(false, a, false, b, (short)0, c, false, false);
}
__device__ __forceinline__ int pk16(float a, float b) {
  h2 v; v[0] = (_Float16)a; v[1] = (_Float16)b;
  return __builtin_bit_cast(int, v);
}

// ---------------- Prep: swizzle weights/biases into WMMA lane layouts ----------------
__global__ __launch_bounds__(256) void gru_prep(const float* __restrict__ w_ih,
                                                const float* __restrict__ w_hh,
                                                const float* __restrict__ b_ih,
                                                const float* __restrict__ b_hh,
                                                unsigned char* __restrict__ ws) {
  int i = blockIdx.x * 256 + threadIdx.x;
  _Float16* whhA = (_Float16*)(ws + WHH_OFF);
  float*    wihA = (float*)(ws + WIH_OFF);
  float*    bC   = (float*)(ws + BIASC_OFF);
  float*    bN   = (float*)(ws + BIASN_OFF);
  float*    wR   = (float*)(ws + WIHR_OFF);

  if (i < 12288) {                       // w_hh f16 A-tiles (16x32 each)
    int m = i >> 10, c = (i >> 9) & 1, l = (i >> 4) & 31;
    int pe = i & 15, p = pe >> 1, e = pe & 1;
    int kl;
    if (l < 16) kl = 2 * p + e;
    else        kl = 16 + 2 * p + e;
    int M = 16 * m + (l & 15);
    int K = 32 * c + kl;
    whhA[i] = (_Float16)w_hh[M * 64 + K];
  } else if (i < 12288 + 1536) {         // w_ih f32 A-tiles (16x4 each)
    int j = i - 12288;
    int fi = j >> 1, comp = j & 1;
    int m = fi >> 5, l = fi & 31;
    int K = ((l < 16) ? 0 : 2) + comp;
    int M = 16 * m + (l & 15);
    wihA[j] = w_ih[M * 4 + K];
  } else if (i < 12288 + 1536 + 3072) {  // C-init biases per D-tile lane layout
    int j = i - (12288 + 1536);
    int m = j >> 8, r = j & 255, l = r >> 3, jj = r & 7;
    int row = 16 * m + jj + ((l < 16) ? 0 : 8);
    bC[j] = (m < 8) ? (b_ih[row] + b_hh[row]) : b_ih[row];
  } else if (i < 12288 + 1536 + 3072 + 1024) {  // b_hh for hn accumulator
    int j = i - (12288 + 1536 + 3072);
    int q = j >> 8, r = j & 255, l = r >> 3, jj = r & 7;
    int row = 128 + 16 * q + jj + ((l < 16) ? 0 : 8);
    bN[j] = b_hh[row];
  } else if (i < 12288 + 1536 + 3072 + 1024 + 768) {  // raw w_ih copy
    int j = i - (12288 + 1536 + 3072 + 1024);
    wR[j] = w_ih[j];
  }
}

// ---------------- Main: 1 wave == 16 batch rows, h kept in VGPRs for all 512 steps ----------------
__global__ __launch_bounds__(256) void gru_main(const float* __restrict__ x,
                                                const unsigned char* __restrict__ ws,
                                                float* __restrict__ out) {
  __shared__ __attribute__((aligned(64))) float s_raw[WS_DW];
  {
    const float* wsf = (const float*)ws;
    for (int i = threadIdx.x; i < WS_DW; i += 256) s_raw[i] = wsf[i];
  }
  __syncthreads();

  const char* sb = (const char*)s_raw;
  const h16* whhv = (const h16*)(sb + WHH_OFF);   // [(m*2+c)*32 + lane]
  const f8*  bcv  = (const f8*)(sb + BIASC_OFF);  // [m*32 + lane]
  const f8*  bnv  = (const f8*)(sb + BIASN_OFF);  // [k*32 + lane]
#if HAVE_X4
  const f2*  wihv = (const f2*)(sb + WIH_OFF);    // [m*32 + lane]
#else
  const float* sWihR = (const float*)(sb + WIHR_OFF);
#endif

  const int lane  = threadIdx.x & 31;
  const int wave  = threadIdx.x >> 5;
  const int bbase = (blockIdx.x * 8 + wave) * 16;
  const int brow  = bbase + (lane & 15);
  const bool lo   = (lane < 16);

  const float* xrow = x + (long)brow * (T_STEPS * 4);

  // h state: f32 copy in D-layout (hF), f16 copy in B-layout (hB0/hB1)
  float hF[4][8];
#pragma unroll
  for (int k = 0; k < 4; ++k)
#pragma unroll
    for (int j = 0; j < 8; ++j) hF[k][j] = 0.0f;
  i8v zi;
#pragma unroll
  for (int q = 0; q < 8; ++q) zi[q] = 0;
  h16 hB0 = __builtin_bit_cast(h16, zi);
  h16 hB1 = __builtin_bit_cast(h16, zi);

  for (int t = 0; t < T_STEPS; ++t) {
    // Keep weights/biases in LDS: block LICM from hoisting (and then spilling)
    // the loop-invariant LDS loads. Register state (hF, hB) is SSA and unaffected.
    asm volatile("" ::: "memory");

    f4 xr4 = *(const f4*)(xrow + t * 4);
#if HAVE_X4
    f2 xB;                                  // B-matrix f32 4x16: lanes<16 hold K0,K1; lanes>=16 hold K2,K3
    xB[0] = lo ? xr4[0] : xr4[2];
    xB[1] = lo ? xr4[1] : xr4[3];
#endif
    auto GX = [&](int m, f8 acc) -> f8 {    // acc += w_ih[tile m] * x_t.T
#if HAVE_X4
      return __builtin_amdgcn_wmma_f32_16x16x4_f32(false, wihv[m * 32 + lane],
                                                   false, xB, (short)0, acc, false, false);
#else
#pragma unroll
      for (int j = 0; j < 8; ++j) {
        int row = 16 * m + j + (lo ? 0 : 8);
        const float* w = sWihR + row * 4;
        acc[j] += w[0] * xr4[0] + w[1] * xr4[1] + w[2] * xr4[2] + w[3] * xr4[3];
      }
      return acc;
#endif
    };

    int P[4][4];
#pragma unroll
    for (int k = 0; k < 4; ++k) {
      // r gate (rows 16k..16k+15)
      f8 cr = bcv[k * 32 + lane];
      cr = GX(k, cr);
      cr = wmma_h(whhv[(2 * k + 0) * 32 + lane], hB0, cr);
      cr = wmma_h(whhv[(2 * k + 1) * 32 + lane], hB1, cr);
      // z gate
      f8 cz = bcv[(4 + k) * 32 + lane];
      cz = GX(4 + k, cz);
      cz = wmma_h(whhv[(2 * (4 + k) + 0) * 32 + lane], hB0, cz);
      cz = wmma_h(whhv[(2 * (4 + k) + 1) * 32 + lane], hB1, cz);
      // n gate: x-part (with b_in) and h-part (with b_hn) kept separate
      f8 cxn = bcv[(8 + k) * 32 + lane];
      cxn = GX(8 + k, cxn);
      f8 chn = bnv[k * 32 + lane];
      chn = wmma_h(whhv[(2 * (8 + k) + 0) * 32 + lane], hB0, chn);
      chn = wmma_h(whhv[(2 * (8 + k) + 1) * 32 + lane], hB1, chn);

#pragma unroll
      for (int j = 0; j < 8; ++j) {
        float r  = fast_sigmoid(cr[j]);
        float z  = fast_sigmoid(cz[j]);
        float n  = fast_tanh(cxn[j] + r * chn[j]);
        float hn = n + z * (hF[k][j] - n);   // (1-z)*n + z*h
        hF[k][j] = hn;
      }
#pragma unroll
      for (int q = 0; q < 4; ++q) P[k][q] = pk16(hF[k][2 * q], hF[k][2 * q + 1]);
    }

    // Re-pack h'.T (D-layout) into next-step B-layout: only a lane-half swap needed.
    i8v bd0, bd1;
#pragma unroll
    for (int c = 0; c < 2; ++c) {
      i8v bd;
#pragma unroll
      for (int q = 0; q < 4; ++q) {
        int other = __shfl_xor(P[2 * c + 1][q], 16, 32);
        bd[q] = lo ? P[2 * c][q] : other;
      }
#pragma unroll
      for (int q = 4; q < 8; ++q) {
        int other = __shfl_xor(P[2 * c][q - 4], 16, 32);
        bd[q] = lo ? other : P[2 * c + 1][q - 4];
      }
      if (c == 0) bd0 = bd; else bd1 = bd;
    }
    hB0 = __builtin_bit_cast(h16, bd0);
    hB1 = __builtin_bit_cast(h16, bd1);
  }

  // Store final h (f32, full precision path) : out[b][d]
  float* op = out + (long)brow * 64 + (lo ? 0 : 8);
#pragma unroll
  for (int k = 0; k < 4; ++k) {
    f4 v0, v1;
#pragma unroll
    for (int j = 0; j < 4; ++j) { v0[j] = hF[k][j]; v1[j] = hF[k][4 + j]; }
    *(f4*)(op + 16 * k)     = v0;
    *(f4*)(op + 16 * k + 4) = v1;
  }
}

extern "C" void kernel_launch(void* const* d_in, const int* in_sizes, int n_in,
                              void* d_out, int out_size, void* d_ws, size_t ws_size,
                              hipStream_t stream) {
  (void)n_in; (void)out_size; (void)ws_size;
  const float* x    = (const float*)d_in[0];
  const float* w_ih = (const float*)d_in[1];
  const float* w_hh = (const float*)d_in[2];
  const float* b_ih = (const float*)d_in[3];
  const float* b_hh = (const float*)d_in[4];
  float* out = (float*)d_out;
  unsigned char* ws = (unsigned char*)d_ws;

  int B = in_sizes[0] / (T_STEPS * 4);        // 16384
  gru_prep<<<73, 256, 0, stream>>>(w_ih, w_hh, b_ih, b_hh, ws);
  gru_main<<<B / 128, 256, 0, stream>>>(x, ws, out);
}
